// MaskedMultiHeadSelfAttention_59090160058956
// MI455X (gfx1250) — matrix operand order repaired and checked by reading, not compiled, not run on hardware
//
#include <hip/hip_runtime.h>

// ---------------------------------------------------------------------------
// MI455X (gfx1250) masked multi-head self-attention
//   bf16 WMMA (v_wmma_f32_16x16x32_bf16) for all matmuls, fp32 accumulate,
//   fp32 softmax, flash-attention streaming over key tiles.
//   Tile staging via double-buffered Tensor Data Mover DMA (tensor_load_to_lds
//   with hardware LDS padding -> 72-half conflict-free pitch), overlapped with
//   WMMA compute via s_wait_tensorcnt <= 2.
// ---------------------------------------------------------------------------

#define BATCH 4
#define SEQ   2048
#define DMODEL 1024
#define NHEAD 16
#define DH    64
#define MTOT  (BATCH * SEQ)      // 8192 tokens

typedef __attribute__((ext_vector_type(16))) __bf16 v16bf;
typedef __attribute__((ext_vector_type(8)))  float  v8f;
typedef __attribute__((ext_vector_type(8)))  unsigned int v8u;
typedef __attribute__((ext_vector_type(4)))  unsigned int v4u;
typedef __attribute__((ext_vector_type(8)))  int v8i;
typedef __attribute__((ext_vector_type(4)))  int v4i;

#if __has_builtin(__builtin_amdgcn_tensor_load_to_lds)
#define HAVE_TDM 1
#if __has_include(<hip/amd_detail/amd_gfx1250_TDM.h>)
#define TDM_SIX_ARGS 1
#endif
#endif

__device__ __forceinline__ unsigned short f2bf(float f) {
    unsigned int u = __float_as_uint(f);
    unsigned int r = u + 0x7FFFu + ((u >> 16) & 1u);   // round-to-nearest-even
    return (unsigned short)(r >> 16);
}

__device__ __forceinline__ v8f wmma_bf16(v16bf a, v16bf b, v8f c) {
    // (neg_a, A, neg_b, B, c_mod, C, reuse_a, reuse_b)
    return __builtin_amdgcn_wmma_f32_16x16x32_bf16(false, a, false, b,
                                                   (short)0, c, false, false);
}

// A fragment: 16x32 bf16, row-major LDS tile with pitch `pitch` (halfs, even).
__device__ __forceinline__ v16bf load_frag_a(const unsigned short* lds,
                                             int row0, int k0, int pitch, int lane) {
    const int m = lane & 15, hl = lane >> 4;
    const unsigned int* rp = (const unsigned int*)(lds + (row0 + m) * pitch + k0);
    v8u u;
#pragma unroll
    for (int j = 0; j < 8; ++j) {
        u[j]     = rp[hl * 8 + j];        // K = hl*16 + 2j
    }
    return __builtin_bit_cast(v16bf, u);
}

// B fragment: 32x16 bf16 from an N-major LDS tile (row n holds K contiguous).
__device__ __forceinline__ v16bf load_frag_b(const unsigned short* lds,
                                             int n0, int k0, int pitch, int lane) {
    const int n = lane & 15, hl = lane >> 4;
    const unsigned int* rp = (const unsigned int*)(lds + (n0 + n) * pitch + k0);
    v8u u;
#pragma unroll
    for (int j = 0; j < 8; ++j) u[j] = rp[hl * 8 + j];
    return __builtin_bit_cast(v16bf, u);
}

// B fragment: 32x16 bf16 from a K-major (row = K) LDS tile: B[k][n] at lds[k*pitch+n].
// 16-bit gathers packed per lane; used for V kept in natural [key][d] order.
__device__ __forceinline__ v16bf load_frag_b_rowmajor(const unsigned short* lds,
                                                      int n0, int k0, int pitch, int lane) {
    const int n = lane & 15, hl = lane >> 4;
    unsigned short tmp[16];
#pragma unroll
    for (int j = 0; j < 16; ++j)
        tmp[j] = lds[(k0 + hl * 16 + j) * pitch + n0 + n];   // K = hl*16 + j
    v8u u;
#pragma unroll
    for (int j = 0; j < 8; ++j)
        u[j] = (unsigned)tmp[2 * j] | ((unsigned)tmp[2 * j + 1] << 16);
    return __builtin_bit_cast(v16bf, u);
}

#if defined(HAVE_TDM)
// LDS byte offset of a __shared__ object: low 32 bits of the generic pointer
// (ISA: LDS aperture addresses truncate to LDS_ADDR.U32).
__device__ __forceinline__ unsigned lds_off(const void* p) {
    return (unsigned)(size_t)p;
}

// 2D bf16 tile load via Tensor Data Mover.
//   rows x cols (elements), global row stride in elements.
//   LDS padding: pad_interval code 4 => every 32 dwords (64 halfs),
//                pad_amount  code 3 => +4 dwords (8 halfs)  -> pitch 72 halfs.
__device__ __forceinline__ void tdm_load_tile_2d(const void* gptr, unsigned ldsbyte,
                                                 int rows, int cols,
                                                 long long row_stride_elems) {
    const unsigned long long ga = (unsigned long long)(size_t)gptr;
    v4u g0;
    g0.x = 1u;                                           // count=1, valid user D#
    g0.y = ldsbyte;                                      // lds_addr (bytes)
    g0.z = (unsigned)(ga & 0xFFFFFFFFu);                 // global_addr[31:0]
    g0.w = (unsigned)((ga >> 32) & 0x01FFFFFFu) | (2u << 30);  // addr[56:32], type=2
    const unsigned td0 = 1u << 30, td1 = 1u << 30;       // huge dims: no OOB clip
    const unsigned long long st0 = (unsigned long long)row_stride_elems;
    v8i g1;
    g1[0] = (int)((1u << 16)            // data_size = 1 -> 2-byte elements
                  | (1u << 20)          // pad_enable
                  | (4u << 22)          // pad_interval code 4 = 32 dwords
                  | (3u << 25));        // pad_amount  code 3 = 4 dwords
    g1[1] = (int)((td0 & 0xFFFFu) << 16);                         // tensor_dim0.lo
    g1[2] = (int)(((td0 >> 16) & 0xFFFFu) | ((td1 & 0xFFFFu) << 16));
    g1[3] = (int)(((td1 >> 16) & 0xFFFFu) | ((unsigned)cols << 16)); // tile_dim0
    g1[4] = (int)(unsigned)rows;                                  // tile_dim1 (dim2=0)
    g1[5] = (int)(unsigned)(st0 & 0xFFFFFFFFu);                   // dim0_stride lo32
    g1[6] = (int)(unsigned)((st0 >> 32) & 0xFFFFu);               // dim0_stride hi16
    g1[7] = 0;
    const v4i z4 = {0, 0, 0, 0};
#if defined(TDM_SIX_ARGS)
    const v8i z8 = {0, 0, 0, 0, 0, 0, 0, 0};
    __builtin_amdgcn_tensor_load_to_lds(g0, g1, z4, z4, z8, 0);
#else
    __builtin_amdgcn_tensor_load_to_lds(g0, g1, z4, z4, 0);
#endif
}

__device__ __forceinline__ void tdm_wait0() {
#if __has_builtin(__builtin_amdgcn_s_wait_tensorcnt)
    __builtin_amdgcn_s_wait_tensorcnt(0);
#else
    asm volatile("s_wait_tensorcnt 0x0" ::: "memory");
#endif
}
__device__ __forceinline__ void tdm_wait2() {   // allow next tile's 2 ops in flight
#if __has_builtin(__builtin_amdgcn_s_wait_tensorcnt)
    __builtin_amdgcn_s_wait_tensorcnt(2);
#else
    asm volatile("s_wait_tensorcnt 0x2" ::: "memory");
#endif
}
#endif  // HAVE_TDM

// ---------------------------------------------------------------------------
// fp32 -> bf16 convert
// ---------------------------------------------------------------------------
__global__ void cvt_f32_bf16(const float* __restrict__ in,
                             unsigned short* __restrict__ out, int n) {
    int i = blockIdx.x * blockDim.x + threadIdx.x;
    int stride = gridDim.x * blockDim.x;
    for (; i < n; i += stride) out[i] = f2bf(in[i]);
}

// ---------------------------------------------------------------------------
// Shared GEMM mainloop: C[128x64] tile, 8 waves, wave = 32x32 (2x2 frags).
// K-step 64; double-buffered TDM pipeline (issue tile k+1, wait tensorcnt<=2,
// compute tile k while tile k+1 streams into the other LDS buffer).
// Ag: [M][K] bf16 row-major.  Bg: [N][K] bf16 row-major (weight (out,in)).
// a_lds: [2][128*72], bt_lds: [2][64*72]
// ---------------------------------------------------------------------------
__device__ __forceinline__ void gemm_mainloop(const unsigned short* __restrict__ Ag,
                                              const unsigned short* __restrict__ Bg,
                                              int K, int M0, int N0,
                                              unsigned short* a_lds,
                                              unsigned short* bt_lds,
                                              v8f acc[2][2]) {
    const int tid  = threadIdx.x;
    const int lane = tid & 31;
    const int w    = tid >> 5;
    const int wm   = (w & 3) * 32;
    const int wn   = (w >> 2) * 32;
    const int nk   = K >> 6;                 // K-steps of 64

#if defined(HAVE_TDM)
    if (w == 0) {                            // prologue: tile 0 -> buffer 0
        tdm_load_tile_2d(Ag + (size_t)M0 * K, lds_off(a_lds), 128, 64, K);
        tdm_load_tile_2d(Bg + (size_t)N0 * K, lds_off(bt_lds), 64, 64, K);
    }
#endif
    for (int ks = 0; ks < nk; ++ks) {
        const int k0 = ks << 6;
#if defined(HAVE_TDM)
        const int buf = ks & 1;
        unsigned short* ab = a_lds  + buf * (128 * 72);
        unsigned short* bb = bt_lds + buf * (64 * 72);
        __syncthreads();                     // all waves done with buf^1
        if (w == 0) {
            if (ks + 1 < nk) {
                unsigned short* an = a_lds  + (buf ^ 1) * (128 * 72);
                unsigned short* bn = bt_lds + (buf ^ 1) * (64 * 72);
                tdm_load_tile_2d(Ag + (size_t)M0 * K + k0 + 64, lds_off(an), 128, 64, K);
                tdm_load_tile_2d(Bg + (size_t)N0 * K + k0 + 64, lds_off(bn), 64, 64, K);
                tdm_wait2();                 // tile ks retired; ks+1 in flight
            } else {
                tdm_wait0();
            }
        }
        __syncthreads();
#else
        unsigned short* ab = a_lds;
        unsigned short* bb = bt_lds;
        __syncthreads();
#pragma unroll
        for (int i = 0; i < 4; ++i) {
            const int idx4 = tid + i * 256;
            const int row = idx4 >> 3, c = (idx4 & 7) * 8;
            *(uint4*)&ab[row * 72 + c] =
                *(const uint4*)(Ag + (size_t)(M0 + row) * K + k0 + c);
        }
#pragma unroll
        for (int i = 0; i < 2; ++i) {
            const int idx4 = tid + i * 256;
            const int row = idx4 >> 3, c = (idx4 & 7) * 8;
            *(uint4*)&bb[row * 72 + c] =
                *(const uint4*)(Bg + (size_t)(N0 + row) * K + k0 + c);
        }
        if (ks + 1 < nk) {
            __builtin_prefetch(Ag + (size_t)(M0 + (tid >> 1)) * K + k0 + 64, 0, 3);
            __builtin_prefetch(Bg + (size_t)(N0 + (tid >> 2)) * K + k0 + 64, 0, 3);
        }
        __syncthreads();
#endif
#pragma unroll
        for (int ksub = 0; ksub < 64; ksub += 32) {
            v16bf a0 = load_frag_a(ab, wm,      ksub, 72, lane);
            v16bf a1 = load_frag_a(ab, wm + 16, ksub, 72, lane);
            v16bf b0 = load_frag_b(bb, wn,      ksub, 72, lane);
            v16bf b1 = load_frag_b(bb, wn + 16, ksub, 72, lane);
            acc[0][0] = wmma_bf16(a0, b0, acc[0][0]);
            acc[0][1] = wmma_bf16(a0, b1, acc[0][1]);
            acc[1][0] = wmma_bf16(a1, b0, acc[1][0]);
            acc[1][1] = wmma_bf16(a1, b1, acc[1][1]);
        }
    }
}

// ---------------------------------------------------------------------------
// Kernel 1: QKV projection -> Q/K/V workspaces [B][H][S][DH] bf16.
// grid = (3072/64, 8192/128), block = 256
// ---------------------------------------------------------------------------
__global__ void __launch_bounds__(256)
qkv_gemm_kernel(const unsigned short* __restrict__ x_bf,
                const unsigned short* __restrict__ wqkv_bf,
                const float* __restrict__ b_qkv,
                unsigned short* __restrict__ q_bf,
                unsigned short* __restrict__ k_bf,
                unsigned short* __restrict__ v_bf) {
    __shared__ unsigned short a_lds[2 * 128 * 72];
    __shared__ unsigned short bt_lds[2 * 64 * 72];
    const int M0 = blockIdx.y * 128;
    const int N0 = blockIdx.x * 64;
    v8f acc[2][2] = {};
    gemm_mainloop(x_bf, wqkv_bf, DMODEL, M0, N0, a_lds, bt_lds, acc);

    const int tid = threadIdx.x, lane = tid & 31, w = tid >> 5;
    const int wm = (w & 3) * 32, wn = (w >> 2) * 32;
    const int nl = lane & 15, hl = lane >> 4;
#pragma unroll
    for (int jj = 0; jj < 2; ++jj) {
        const int e = N0 + wn + jj * 16 + nl;
        const float bias = b_qkv[e];
        const int part = e >> 10;
        const int rem = e & 1023;
        const int h = rem >> 6, d = rem & 63;
        unsigned short* dst = (part == 0) ? q_bf : (part == 1) ? k_bf : v_bf;
#pragma unroll
        for (int ii = 0; ii < 2; ++ii) {
            v8f cc = acc[ii][jj];
#pragma unroll
            for (int r = 0; r < 8; ++r) {
                const int mrow = M0 + wm + ii * 16 + hl * 8 + r;
                const int b = mrow >> 11, s = mrow & 2047;
                const size_t off = ((size_t)((b * NHEAD + h) * SEQ + s)) * DH + d;
                dst[off] = f2bf(cc[r] + bias);
            }
        }
    }
}

// ---------------------------------------------------------------------------
// Kernel 2: causal flash attention.
// grid = (SEQ/128, BATCH*NHEAD), block = 256 (wave w owns 16 q rows)
// K and V tiles stream through double-buffered TDM; V stays [key][d] row-major
// and its B-operand is gathered with 16-bit LDS loads.
// ---------------------------------------------------------------------------
__global__ void __launch_bounds__(256)
flash_attn_kernel(const unsigned short* __restrict__ Q,
                  const unsigned short* __restrict__ Kk,
                  const unsigned short* __restrict__ V,
                  unsigned short* __restrict__ Ao) {
    __shared__ unsigned short q_lds[128 * 72];
    __shared__ unsigned short k_lds[2 * 64 * 72];    // [key][d]
    __shared__ unsigned short v_lds[2 * 64 * 72];    // [key][d]
    __shared__ unsigned short p_lds[128 * 72];

    const int tid = threadIdx.x, lane = tid & 31, w = tid >> 5;
    const int nl = lane & 15, hl = lane >> 4;
    const int q0 = blockIdx.x * 128;
    const int bh = blockIdx.y;                   // b*16 + h
    const size_t base = (size_t)bh * SEQ * DH;
    const int nkt = (q0 >> 6) + 2;               // causal: keys <= q0+127

#if defined(HAVE_TDM)
    if (w == 0) {                                // prologue: Q + tile 0
        tdm_load_tile_2d(Q + base + (size_t)q0 * DH, lds_off(q_lds), 128, 64, DH);
        tdm_load_tile_2d(Kk + base, lds_off(k_lds), 64, 64, DH);
        tdm_load_tile_2d(V + base, lds_off(v_lds), 64, 64, DH);
    }
#else
#pragma unroll
    for (int i = 0; i < 4; ++i) {
        const int idx4 = tid + i * 256;
        const int row = idx4 >> 3, c = (idx4 & 7) * 8;
        *(uint4*)&q_lds[row * 72 + c] =
            *(const uint4*)(Q + base + (size_t)(q0 + row) * DH + c);
    }
#endif

    v8f o[4] = {};
    float mrow[8], lrow[8];
#pragma unroll
    for (int r = 0; r < 8; ++r) { mrow[r] = -3.4e38f; lrow[r] = 0.f; }

    for (int kt = 0; kt < nkt; ++kt) {
        const int key0 = kt * 64;
#if defined(HAVE_TDM)
        const int buf = kt & 1;
        unsigned short* kb = k_lds + buf * (64 * 72);
        unsigned short* vb = v_lds + buf * (64 * 72);
        __syncthreads();                         // all waves done with buf^1
        if (w == 0) {
            if (kt + 1 < nkt) {
                unsigned short* kn = k_lds + (buf ^ 1) * (64 * 72);
                unsigned short* vn = v_lds + (buf ^ 1) * (64 * 72);
                tdm_load_tile_2d(Kk + base + (size_t)(key0 + 64) * DH, lds_off(kn), 64, 64, DH);
                tdm_load_tile_2d(V  + base + (size_t)(key0 + 64) * DH, lds_off(vn), 64, 64, DH);
                tdm_wait2();                     // current K/V retired; next in flight
            } else {
                tdm_wait0();
            }
        }
        __syncthreads();
#else
        unsigned short* kb = k_lds;
        unsigned short* vb = v_lds;
        __syncthreads();
#pragma unroll
        for (int i = 0; i < 2; ++i) {
            const int idx4 = tid + i * 256;
            const int row = idx4 >> 3, c = (idx4 & 7) * 8;
            *(uint4*)&kb[row * 72 + c] =
                *(const uint4*)(Kk + base + (size_t)(key0 + row) * DH + c);
            *(uint4*)&vb[row * 72 + c] =
                *(const uint4*)(V + base + (size_t)(key0 + row) * DH + c);
        }
        __syncthreads();
#endif

        // scores: S = Q * K^T   (K tile is key-major -> N-major for B operand)
        v8f sc[4] = {};
#pragma unroll
        for (int ks = 0; ks < 2; ++ks) {
            v16bf a = load_frag_a(q_lds, w * 16, ks * 32, 72, lane);
#pragma unroll
            for (int jj = 0; jj < 4; ++jj) {
                v16bf b = load_frag_b(kb, jj * 16, ks * 32, 72, lane);
                sc[jj] = wmma_bf16(a, b, sc[jj]);
            }
        }

        // causal mask + online softmax (rows live on 16-lane halves)
        const float scale = 0.125f;              // 1/sqrt(64)
#pragma unroll
        for (int r = 0; r < 8; ++r) {
            const int qg = q0 + w * 16 + hl * 8 + r;
            float v4[4];
            float rmax = -3.4e38f;
#pragma unroll
            for (int jj = 0; jj < 4; ++jj) {
                const int kg = key0 + jj * 16 + nl;
                float s = sc[jj][r] * scale;
                if (kg > qg) s = -3.4e38f;
                v4[jj] = s;
                rmax = fmaxf(rmax, s);
            }
#pragma unroll
            for (int off = 8; off >= 1; off >>= 1)
                rmax = fmaxf(rmax, __shfl_xor(rmax, off, 32));
            const float newm = fmaxf(mrow[r], rmax);
            const float alpha = __expf(mrow[r] - newm);
            float rsum = 0.f;
#pragma unroll
            for (int jj = 0; jj < 4; ++jj) {
                const float p = __expf(v4[jj] - newm);
                rsum += p;
                p_lds[(w * 16 + hl * 8 + r) * 72 + jj * 16 + nl] = f2bf(p);
            }
#pragma unroll
            for (int off = 8; off >= 1; off >>= 1)
                rsum += __shfl_xor(rsum, off, 32);
            lrow[r] = lrow[r] * alpha + rsum;
            mrow[r] = newm;
#pragma unroll
            for (int jj = 0; jj < 4; ++jj) o[jj][r] *= alpha;
        }

        // O += P * V   (p rows are per-wave private; V is K-major [key][d])
#pragma unroll
        for (int ks = 0; ks < 2; ++ks) {
            v16bf a = load_frag_a(p_lds, w * 16, ks * 32, 72, lane);
#pragma unroll
            for (int jj = 0; jj < 4; ++jj) {
                v16bf b = load_frag_b_rowmajor(vb, jj * 16, ks * 32, 72, lane);
                o[jj] = wmma_bf16(a, b, o[jj]);
            }
        }
    }

    // epilogue: normalize, store attn_bf [B][S][DMODEL]
    const int b = bh >> 4, h = bh & 15;
#pragma unroll
    for (int r = 0; r < 8; ++r) {
        const float inv = 1.0f / lrow[r];
        const int s = q0 + w * 16 + hl * 8 + r;
        const size_t rowoff = ((size_t)(b * SEQ + s)) * DMODEL + h * DH;
#pragma unroll
        for (int jj = 0; jj < 4; ++jj)
            Ao[rowoff + jj * 16 + nl] = f2bf(o[jj][r] * inv);
    }
}

// ---------------------------------------------------------------------------
// Kernel 3: output projection + bias, fp32 out.
// grid = (1024/64, 8192/128), block = 256.
// ---------------------------------------------------------------------------
__global__ void __launch_bounds__(256)
oproj_gemm_kernel(const unsigned short* __restrict__ a_bf,
                  const unsigned short* __restrict__ wo_bf,
                  const float* __restrict__ b_o,
                  float* __restrict__ out) {
    __shared__ unsigned short a_lds[2 * 128 * 72];
    __shared__ unsigned short bt_lds[2 * 64 * 72];
    const int M0 = blockIdx.y * 128;
    const int N0 = blockIdx.x * 64;
    v8f acc[2][2] = {};
    gemm_mainloop(a_bf, wo_bf, DMODEL, M0, N0, a_lds, bt_lds, acc);

    const int tid = threadIdx.x, lane = tid & 31, w = tid >> 5;
    const int wm = (w & 3) * 32, wn = (w >> 2) * 32;
    const int nl = lane & 15, hl = lane >> 4;
#pragma unroll
    for (int jj = 0; jj < 2; ++jj) {
        const int e = N0 + wn + jj * 16 + nl;
        const float bias = b_o[e];
#pragma unroll
        for (int ii = 0; ii < 2; ++ii) {
            v8f cc = acc[ii][jj];
#pragma unroll
            for (int r = 0; r < 8; ++r) {
                const int mrow = M0 + wm + ii * 16 + hl * 8 + r;
                out[(size_t)mrow * DMODEL + e] = cc[r] + bias;
            }
        }
    }
}

// ---------------------------------------------------------------------------
// Launch
// ---------------------------------------------------------------------------
extern "C" void kernel_launch(void* const* d_in, const int* in_sizes, int n_in,
                              void* d_out, int out_size, void* d_ws, size_t ws_size,
                              hipStream_t stream) {
    const float* x     = (const float*)d_in[0];   // [4,2048,1024]
    const float* Wqkv  = (const float*)d_in[1];   // [3072,1024]
    const float* bqkv  = (const float*)d_in[2];   // [3072]
    const float* Wo    = (const float*)d_in[3];   // [1024,1024]
    const float* bo    = (const float*)d_in[4];   // [1024]
    float* out = (float*)d_out;

    unsigned short* ws = (unsigned short*)d_ws;
    const size_t N_X  = (size_t)MTOT * DMODEL;        // 8388608
    const size_t N_WQ = (size_t)3 * DMODEL * DMODEL;  // 3145728
    const size_t N_WO = (size_t)DMODEL * DMODEL;      // 1048576
    unsigned short* x_bf    = ws;
    unsigned short* wqkv_bf = x_bf + N_X;
    unsigned short* wo_bf   = wqkv_bf + N_WQ;
    unsigned short* q_bf    = wo_bf + N_WO;
    unsigned short* k_bf    = q_bf + N_X;
    unsigned short* v_bf    = k_bf + N_X;
    unsigned short* a_bf    = v_bf + N_X;
    // total ws: ~92.3 MB

    // 0) precision convert
    cvt_f32_bf16<<<4096, 256, 0, stream>>>(x, x_bf, (int)N_X);
    cvt_f32_bf16<<<4096, 256, 0, stream>>>(Wqkv, wqkv_bf, (int)N_WQ);
    cvt_f32_bf16<<<2048, 256, 0, stream>>>(Wo, wo_bf, (int)N_WO);

    // 1) QKV projection
    qkv_gemm_kernel<<<dim3(3 * DMODEL / 64, MTOT / 128), 256, 0, stream>>>(
        x_bf, wqkv_bf, bqkv, q_bf, k_bf, v_bf);

    // 2) causal flash attention
    flash_attn_kernel<<<dim3(SEQ / 128, BATCH * NHEAD), 256, 0, stream>>>(
        q_bf, k_bf, v_bf, a_bf);

    // 3) output projection
    oproj_gemm_kernel<<<dim3(DMODEL / 64, MTOT / 128), 256, 0, stream>>>(
        a_bf, wo_bf, bo, out);
}
